// TopKastLinear_44581760532972
// MI455X (gfx1250) — matrix operand order repaired and checked by reading, not compiled, not run on hardware
//
#include <hip/hip_runtime.h>

// ---------------------------------------------------------------------------
// Top-KAST linear forward for MI455X (gfx1250, wave32, WMMA, async-LDS).
//   inputs  d_in[0]: float32 [4096, 8192]   (X)
//   weight  d_in[1]: float32 [8192, 8192]   (W, row-major [out, in])
//   bias    d_in[2]: float32 [8192]
//   out     d_out  : float32 [4096, 8192]
// ---------------------------------------------------------------------------

#define BATCH 4096
#define IN_F  8192
#define OUT_F 8192
#define NW    (OUT_F * IN_F)            // 67108864 weights
#define NW4   (NW / 4)
#define NX4   ((BATCH * IN_F) / 4)
#define K_RANK 60397976u                 // floor(0.9 * (NW - 1))
#define K_FRAC 0.7f                      // frac(0.9 * (NW - 1))

typedef __attribute__((ext_vector_type(16))) __bf16 v16bf;
typedef __attribute__((ext_vector_type(8)))  float  v8f;
typedef int v4i __attribute__((vector_size(16)));   // matches async builtin param

struct SelState {
    unsigned int prefix;     // selected high bits of the k-th |w| bit pattern
    unsigned int rank;       // remaining zero-based rank within matched set
    unsigned int count_eq;   // multiplicity of the selected value
    unsigned int next_min;   // min bit pattern strictly greater than value
    float        threshold;  // final interpolated quantile
    unsigned int hist[256];
};

union FragU { uint4 u[2]; v16bf v; };

__device__ __forceinline__ unsigned short f32_to_bf16_rne(float f) {
    unsigned int u = __float_as_uint(f);
    unsigned int r = u + 0x7FFFu + ((u >> 16) & 1u);
    return (unsigned short)(r >> 16);
}

// ---------------- gfx1250 async global->LDS (ASYNCcnt) helpers -------------

#if __has_builtin(__builtin_amdgcn_global_load_async_to_lds_b128)
#define HAVE_ASYNC_LDS 1
#else
#define HAVE_ASYNC_LDS 0
#endif

__device__ __forceinline__ void async_copy_b128(const unsigned short* g,
                                                unsigned short* l) {
#if HAVE_ASYNC_LDS
    __builtin_amdgcn_global_load_async_to_lds_b128(
        (__attribute__((address_space(1))) v4i*)g,
        (__attribute__((address_space(3))) v4i*)l,
        0, 0);
#else
    *(uint4*)l = *(const uint4*)g;   // sync fallback (VGPR-staged)
#endif
}

__device__ __forceinline__ void wait_async_le8() {
#if __has_builtin(__builtin_amdgcn_s_wait_asynccnt)
    __builtin_amdgcn_s_wait_asynccnt(8);
#else
    asm volatile("s_wait_asynccnt 0x8" ::: "memory");
#endif
}

__device__ __forceinline__ void wait_async_le0() {
#if __has_builtin(__builtin_amdgcn_s_wait_asynccnt)
    __builtin_amdgcn_s_wait_asynccnt(0);
#else
    asm volatile("s_wait_asynccnt 0x0" ::: "memory");
#endif
}

// --------------------------- threshold pipeline ----------------------------

__global__ void tk_init(SelState* st) {
    if (threadIdx.x == 0) {
        st->prefix   = 0u;
        st->rank     = K_RANK;
        st->count_eq = 0u;
        st->next_min = 0x7FFFFFFFu;
        st->threshold = 0.0f;
    }
    st->hist[threadIdx.x] = 0u;
}

__global__ void tk_hist(const uint4* __restrict__ wbits, SelState* __restrict__ st,
                        unsigned int shift, unsigned int pmask) {
    __shared__ unsigned int lh[256];
    lh[threadIdx.x] = 0u;
    __syncthreads();
    const unsigned int prefix = st->prefix;
    const int stride = gridDim.x * blockDim.x;
    for (int i = blockIdx.x * blockDim.x + threadIdx.x; i < NW4; i += stride) {
        uint4 v = wbits[i];
        unsigned int b;
        b = v.x & 0x7FFFFFFFu; if ((b & pmask) == prefix) atomicAdd(&lh[(b >> shift) & 0xFFu], 1u);
        b = v.y & 0x7FFFFFFFu; if ((b & pmask) == prefix) atomicAdd(&lh[(b >> shift) & 0xFFu], 1u);
        b = v.z & 0x7FFFFFFFu; if ((b & pmask) == prefix) atomicAdd(&lh[(b >> shift) & 0xFFu], 1u);
        b = v.w & 0x7FFFFFFFu; if ((b & pmask) == prefix) atomicAdd(&lh[(b >> shift) & 0xFFu], 1u);
    }
    __syncthreads();
    if (lh[threadIdx.x]) atomicAdd(&st->hist[threadIdx.x], lh[threadIdx.x]);
}

__global__ void tk_scan(SelState* st, unsigned int shift, int last_round) {
    unsigned long long cum = 0;
    const unsigned int r = st->rank;
    for (int b = 0; b < 256; ++b) {
        unsigned int c = st->hist[b];
        if (cum + c > (unsigned long long)r) {
            st->prefix |= ((unsigned int)b) << shift;
            st->rank = r - (unsigned int)cum;
            if (last_round) st->count_eq = c;
            break;
        }
        cum += c;
    }
    for (int b = 0; b < 256; ++b) st->hist[b] = 0u;
}

__global__ void tk_nextmin(const uint4* __restrict__ wbits, SelState* __restrict__ st) {
    const unsigned int v = st->prefix;
    unsigned int local = 0x7FFFFFFFu;
    const int stride = gridDim.x * blockDim.x;
    for (int i = blockIdx.x * blockDim.x + threadIdx.x; i < NW4; i += stride) {
        uint4 w = wbits[i];
        unsigned int b;
        b = w.x & 0x7FFFFFFFu; if (b > v && b < local) local = b;
        b = w.y & 0x7FFFFFFFu; if (b > v && b < local) local = b;
        b = w.z & 0x7FFFFFFFu; if (b > v && b < local) local = b;
        b = w.w & 0x7FFFFFFFu; if (b > v && b < local) local = b;
    }
    if (local != 0x7FFFFFFFu) atomicMin(&st->next_min, local);
}

__global__ void tk_finalize(SelState* st) {
    const unsigned int v = st->prefix;
    const float a0 = __uint_as_float(v);
    const float a1 = (st->rank + 1u < st->count_eq) ? a0 : __uint_as_float(st->next_min);
    st->threshold = a0 + K_FRAC * (a1 - a0);
}

// ----------------------- mask + convert to bf16 ----------------------------

__global__ void tk_mask_cvt_w(const float4* __restrict__ w,
                              const SelState* __restrict__ st,
                              ushort4* __restrict__ wbf) {
    const float thr = st->threshold;
    const int stride = gridDim.x * blockDim.x;
    for (int i = blockIdx.x * blockDim.x + threadIdx.x; i < NW4; i += stride) {
        float4 v = w[i];
        ushort4 o;
        o.x = (fabsf(v.x) >= thr) ? f32_to_bf16_rne(v.x) : (unsigned short)0;
        o.y = (fabsf(v.y) >= thr) ? f32_to_bf16_rne(v.y) : (unsigned short)0;
        o.z = (fabsf(v.z) >= thr) ? f32_to_bf16_rne(v.z) : (unsigned short)0;
        o.w = (fabsf(v.w) >= thr) ? f32_to_bf16_rne(v.w) : (unsigned short)0;
        wbf[i] = o;
    }
}

__global__ void tk_cvt_x(const float4* __restrict__ x, ushort4* __restrict__ xbf) {
    const int stride = gridDim.x * blockDim.x;
    for (int i = blockIdx.x * blockDim.x + threadIdx.x; i < NX4; i += stride) {
        float4 v = x[i];
        ushort4 o;
        o.x = f32_to_bf16_rne(v.x);
        o.y = f32_to_bf16_rne(v.y);
        o.z = f32_to_bf16_rne(v.z);
        o.w = f32_to_bf16_rne(v.w);
        xbf[i] = o;
    }
}

// ------------------------------ WMMA GEMM ----------------------------------
// C[m,n] = sum_k Xbf[m,k] * Wbf[n,k] + bias[n]
// Block tile 128(M) x 128(N) x 64(K); 8 waves; wave tile 64(M) x 32(N).
// Double-buffered LDS filled by GLOBAL_LOAD_ASYNC_TO_LDS_B128 (ASYNCcnt);
// 16 v_wmma_f32_16x16x32_bf16 per stage overlap the next stage's async DMA.

#define TILE_M 128
#define TILE_N 128
#define TILE_K 64
#define LDSS   72   // padded K-stride (halves): 144B rows, 16B aligned

__device__ __forceinline__ void tk_stage(const unsigned short* __restrict__ Xbf,
                                         const unsigned short* __restrict__ Wbf,
                                         unsigned short* Abuf, unsigned short* Bbuf,
                                         int blockM, int blockN, int kt, int tid) {
    // 1024 b128 chunks per matrix; 256 threads -> 4 A + 4 B async ops each.
#pragma unroll
    for (int l = 0; l < 4; ++l) {
        const int c   = tid + l * 256;   // 0..1023
        const int row = c >> 3;          // 0..127
        const int cc  = c & 7;           // 0..7 (8 halves each)
        async_copy_b128(Xbf + (size_t)(blockM + row) * IN_F + kt + cc * 8,
                        Abuf + row * LDSS + cc * 8);
        async_copy_b128(Wbf + (size_t)(blockN + row) * IN_F + kt + cc * 8,
                        Bbuf + row * LDSS + cc * 8);
    }
}

__global__ __launch_bounds__(256) void tk_gemm(const unsigned short* __restrict__ Xbf,
                                               const unsigned short* __restrict__ Wbf,
                                               const float* __restrict__ bias,
                                               float* __restrict__ out) {
    __shared__ unsigned short As[2][TILE_M * LDSS];
    __shared__ unsigned short Bs[2][TILE_N * LDSS];

    const int tid  = threadIdx.x;
    const int wave = tid >> 5;          // 0..7
    const int lane = tid & 31;
    const int wm   = wave >> 2;         // 0..1  (M direction)
    const int wn   = wave & 3;          // 0..3  (N direction)
    const int hi   = lane >> 4;         // 0..1
    const int nl   = lane & 15;

    const int blockM = blockIdx.y * TILE_M;
    const int blockN = blockIdx.x * TILE_N;

    v8f acc[4][2];
#pragma unroll
    for (int sm = 0; sm < 4; ++sm)
#pragma unroll
        for (int sn = 0; sn < 2; ++sn)
            acc[sm][sn] = (v8f){};

    // Prologue: fill stage 0.
    tk_stage(Xbf, Wbf, &As[0][0], &Bs[0][0], blockM, blockN, 0, tid);

    for (int kt = 0; kt < IN_F; kt += TILE_K) {
        const int buf = (kt / TILE_K) & 1;
        const bool has_next = (kt + TILE_K) < IN_F;

        // Kick off next stage's DMA into the other buffer, then drain the
        // 8 older async ops that filled the current buffer.
        if (has_next) {
            tk_stage(Xbf, Wbf, &As[buf ^ 1][0], &Bs[buf ^ 1][0],
                     blockM, blockN, kt + TILE_K, tid);
            wait_async_le8();
        } else {
            wait_async_le0();
        }
        __syncthreads();   // current buffer visible to all waves

        const unsigned short* Ab = &As[buf][0];
        const unsigned short* Bb = &Bs[buf][0];
#pragma unroll
        for (int ks = 0; ks < 2; ++ks) {   // two 32-wide K steps per stage
            const int k0 = ks * 32;
            // A frags: lane holds row m=nl; halves 0-7 = K[hi*8..],
            //          halves 8-15 = K[16+hi*8..]
            FragU a[4];
#pragma unroll
            for (int sm = 0; sm < 4; ++sm) {
                const unsigned short* base =
                    Ab + (wm * 64 + sm * 16 + nl) * LDSS + k0;
                a[sm].u[0] = *(const uint4*)(base + hi * 8);
                a[sm].u[1] = *(const uint4*)(base + 16 + hi * 8);
            }
            // B frags: lane holds column n=nl (= W row), halves 0-7 = K[hi*8..],
            //          halves 8-15 = K[16+hi*8..]
            FragU b[2];
#pragma unroll
            for (int sn = 0; sn < 2; ++sn) {
                const unsigned short* base =
                    Bb + (wn * 32 + sn * 16 + nl) * LDSS + k0;
                b[sn].u[0] = *(const uint4*)(base + hi * 8);
                b[sn].u[1] = *(const uint4*)(base + 16 + hi * 8);
            }
#pragma unroll
            for (int sm = 0; sm < 4; ++sm)
#pragma unroll
                for (int sn = 0; sn < 2; ++sn)
                    acc[sm][sn] = __builtin_amdgcn_wmma_f32_16x16x32_bf16(
                        false, a[sm].v, false, b[sn].v, (short)0,
                        acc[sm][sn], false, false);
        }
        __syncthreads();   // all waves done reading buf before it is refilled
    }

    // Epilogue: element (m = r + 8*hi, n = nl) of each 16x16 tile; add bias.
    float bv[2];
#pragma unroll
    for (int sn = 0; sn < 2; ++sn)
        bv[sn] = bias[blockN + wn * 32 + sn * 16 + nl];

#pragma unroll
    for (int sm = 0; sm < 4; ++sm)
#pragma unroll
        for (int sn = 0; sn < 2; ++sn) {
            const int n = blockN + wn * 32 + sn * 16 + nl;
#pragma unroll
            for (int r = 0; r < 8; ++r) {
                const int m = blockM + wm * 64 + sm * 16 + r + 8 * hi;
                out[(size_t)m * OUT_F + n] = acc[sm][sn][r] + bv[sn];
            }
        }
}

// ------------------------------- launcher ----------------------------------

extern "C" void kernel_launch(void* const* d_in, const int* in_sizes, int n_in,
                              void* d_out, int out_size, void* d_ws, size_t ws_size,
                              hipStream_t stream) {
    (void)in_sizes; (void)n_in; (void)out_size; (void)ws_size;

    const float* x    = (const float*)d_in[0];
    const float* w    = (const float*)d_in[1];
    const float* bias = (const float*)d_in[2];
    float*       out  = (float*)d_out;

    char* ws = (char*)d_ws;
    SelState*       st  = (SelState*)ws;
    unsigned short* Wbf = (unsigned short*)(ws + 1024);
    unsigned short* Xbf = (unsigned short*)(ws + 1024 + (size_t)NW * sizeof(unsigned short));

    const uint4*  wbits = (const uint4*)w;
    const float4* wf4   = (const float4*)w;
    const float4* xf4   = (const float4*)x;

    // 1) Exact 0.9-quantile of |W| via 4-round radix select on f32 bit patterns.
    tk_init<<<1, 256, 0, stream>>>(st);
    for (int r = 0; r < 4; ++r) {
        const unsigned int shift = 24u - 8u * (unsigned)r;
        const unsigned int pmask = (r == 0) ? 0u : (0xFFFFFFFFu << (32u - 8u * (unsigned)r));
        tk_hist<<<4096, 256, 0, stream>>>(wbits, st, shift, pmask);
        tk_scan<<<1, 1, 0, stream>>>(st, shift, r == 3);
    }
    tk_nextmin<<<4096, 256, 0, stream>>>(wbits, st);
    tk_finalize<<<1, 1, 0, stream>>>(st);

    // 2) Mask + convert W to bf16 (fits in 192MB L2); convert X to bf16.
    tk_mask_cvt_w<<<4096, 256, 0, stream>>>(wf4, st, (ushort4*)Wbf);
    tk_cvt_x<<<2048, 256, 0, stream>>>(xf4, (ushort4*)Xbf);

    // 3) WMMA bf16 GEMM, async-LDS double-buffered, f32 accumulate + bias.
    dim3 grid(OUT_F / TILE_N, BATCH / TILE_M);
    tk_gemm<<<grid, 256, 0, stream>>>(Xbf, Wbf, bias, out);
}
